// Qwen3MoeSparseMoeBlockOld_46909632807483
// MI455X (gfx1250) — matrix operand order repaired and checked by reading, not compiled, not run on hardware
//
#include <hip/hip_runtime.h>

#define T_TOK 2048
#define H_DIM 2048
#define I_DIM 768
#define N_EXP 64
#define TOPK  8
#define M_TILE 32
#define NWAVE 16

typedef __bf16 bf16;
typedef __bf16 v16bf __attribute__((ext_vector_type(16)));
typedef __bf16 v8bf  __attribute__((ext_vector_type(8)));
typedef __bf16 v4bf  __attribute__((ext_vector_type(4)));
typedef __bf16 v2bf  __attribute__((ext_vector_type(2)));
typedef float  v8f   __attribute__((ext_vector_type(8)));

__device__ __forceinline__ v16bf cat8(v8bf lo, v8bf hi) {
  v16bf r;
#pragma unroll
  for (int i = 0; i < 8; ++i) { r[i] = lo[i]; r[i + 8] = hi[i]; }
  return r;
}

// A fragment (16x32 bf16, MxK), ISA layout: lanes 0-15: M=lane, K {0..7,16..23};
// lanes 16-31: M=lane-16, K offset +8.
__device__ __forceinline__ v16bf load_a_frag(const bf16* base, int stride, int k0, int lane) {
  const int M  = lane & 15;
  const int kb = (lane < 16) ? 0 : 8;
  const bf16* p = base + M * stride + k0 + kb;
  v8bf lo = *(const v8bf*)p;
  v8bf hi = *(const v8bf*)(p + 16);
  return cat8(lo, hi);
}

// B fragment (32x16 bf16, KxN) from LDS tile stored [N][K] (K-contiguous):
// lanes 0-15: N=lane, K {0..7,16..23}; lanes 16-31: N=lane-16, K offset +8.
__device__ __forceinline__ v16bf load_b_frag(const bf16* tile, int lane) {
  const int N  = lane & 15;
  const int kb = (lane < 16) ? 0 : 8;
  const bf16* p = tile + N * 32 + kb;
  v8bf lo = *(const v8bf*)p;
  v8bf hi = *(const v8bf*)(p + 16);
  return cat8(lo, hi);
}

// Stream a 32(K) x 16(N) fp32 tile, convert to bf16 pairs along K (one
// v_cvt_pk_bf16_f32 + one ds_store_b32 per packed word), store [N][K].
// lane -> K-pair (lane&15) x N-half ((lane>>4)*8).
__device__ __forceinline__ void stage_b_tile(const float* g, int ldg, bf16* tile, int lane) {
  const int kp = lane & 15;
  const int nh = (lane >> 4) * 8;
  const float4* r0 = (const float4*)(g + (size_t)(2 * kp)     * ldg + nh);
  const float4* r1 = (const float4*)(g + (size_t)(2 * kp + 1) * ldg + nh);
  float4 a0 = r0[0], a1 = r0[1];
  float4 b0 = r1[0], b1 = r1[1];
  float ra[8] = {a0.x, a0.y, a0.z, a0.w, a1.x, a1.y, a1.z, a1.w};
  float rb[8] = {b0.x, b0.y, b0.z, b0.w, b1.x, b1.y, b1.z, b1.w};
#pragma unroll
  for (int j = 0; j < 8; ++j) {
    v2bf pk; pk[0] = (bf16)ra[j]; pk[1] = (bf16)rb[j];
    *(v2bf*)&tile[(nh + j) * 32 + 2 * kp] = pk;
  }
}

#define WMMA_BF16(A, B, C) \
  __builtin_amdgcn_wmma_f32_16x16x32_bf16(false, (A), false, (B), (short)0, (C), false, false)

// ---------------- Router: logits -> softmax -> top-8 -> per-expert token lists ----
__global__ void router_kernel(const float* __restrict__ x, const float* __restrict__ wr,
                              int* __restrict__ counts, int* __restrict__ tok_ids,
                              float* __restrict__ tok_w) {
  __shared__ float sl[N_EXP];
  const int t = blockIdx.x;
  const int e = threadIdx.x;
  const float* xr = x + (size_t)t * H_DIM;
  float acc = 0.f;
  for (int h = 0; h < H_DIM; ++h)
    acc = fmaf(xr[h], wr[(size_t)h * N_EXP + e], acc);
  sl[e] = acc;
  __syncthreads();
  if (e == 0) {
    float mx = sl[0];
    for (int i = 1; i < N_EXP; ++i) mx = fmaxf(mx, sl[i]);
    float s = 0.f;
    for (int i = 0; i < N_EXP; ++i) { float p = __expf(sl[i] - mx); sl[i] = p; s += p; }
    (void)s; // top-k is monotone in unnormalized probs; renormalize over top-k below
    float topv[TOPK]; int topi[TOPK]; float ts = 0.f;
    for (int k = 0; k < TOPK; ++k) {
      float bv = -1.f; int bi = 0;
      for (int i = 0; i < N_EXP; ++i) if (sl[i] > bv) { bv = sl[i]; bi = i; }
      topv[k] = bv; topi[k] = bi; sl[bi] = -2.f; ts += bv;
    }
    const float inv = 1.f / ts;
    for (int k = 0; k < TOPK; ++k) {
      int ex  = topi[k];
      int pos = atomicAdd(&counts[ex], 1);
      tok_ids[(size_t)ex * T_TOK + pos] = t;
      tok_w [(size_t)ex * T_TOK + pos] = topv[k] * inv;
    }
  }
}

// ---------------- Fused expert FFN: h = silu(x@wg)*(x@wu); out += w * (h@wd) -----
__global__ __launch_bounds__(NWAVE * 32) void moe_ffn_kernel(
    const float* __restrict__ x, const float* __restrict__ w_gate,
    const float* __restrict__ w_up, const float* __restrict__ w_down,
    const int* __restrict__ counts, const int* __restrict__ tok_ids,
    const float* __restrict__ tok_w, float* __restrict__ out) {
  __shared__ bf16 xs[M_TILE][H_DIM];          // 128 KB: 32-token tile of x, bf16
  __shared__ bf16 hs[M_TILE][I_DIM];          //  48 KB: intermediate activations
  __shared__ bf16 bstage[NWAVE][2][16 * 32];  //  32 KB: per-wave B-tile staging
  __shared__ int   sid[M_TILE];
  __shared__ float swt[M_TILE];

  const int e    = blockIdx.y;
  const int tile = blockIdx.x;
  const int cnt  = counts[e];
  if (tile * M_TILE >= cnt) return;

  const int tid  = threadIdx.x;
  const int lane = tid & 31;
  const int wv   = tid >> 5;

  if (tid < M_TILE) {
    int rr = tile * M_TILE + tid;
    if (rr < cnt) { sid[tid] = tok_ids[(size_t)e * T_TOK + rr]; swt[tid] = tok_w[(size_t)e * T_TOK + rr]; }
    else          { sid[tid] = 0; swt[tid] = 0.f; }
  }
  __syncthreads();

  // Stage 0: gather 32 token rows of x into LDS as bf16 (32*512 = 16384 float4s).
  for (int it = 0; it < 32; ++it) {
    int li4 = tid + it * (NWAVE * 32);  // 0..16383
    int row = li4 >> 9;                 // / 512
    int c4  = li4 & 511;
    float4 v = ((const float4*)(x + (size_t)sid[row] * H_DIM))[c4];
    v4bf pk; pk[0] = (bf16)v.x; pk[1] = (bf16)v.y; pk[2] = (bf16)v.z; pk[3] = (bf16)v.w;
    *(v4bf*)&xs[row][c4 * 4] = pk;
  }
  __syncthreads();

  const float* wg_e = w_gate + (size_t)e * H_DIM * I_DIM;
  const float* wu_e = w_up   + (size_t)e * H_DIM * I_DIM;
  const float* wd_e = w_down + (size_t)e * I_DIM * H_DIM;

  // Stage 1: gate/up GEMMs. Each of 16 waves owns 48 of 768 I-columns (3 N-tiles),
  // two M-tiles deep; every staged B fragment feeds 2 WMMAs per matrix.
  v8f accg[3][2], accu[3][2];
#pragma unroll
  for (int ct = 0; ct < 3; ++ct)
#pragma unroll
    for (int m = 0; m < 2; ++m) { accg[ct][m] = {}; accu[ct][m] = {}; }

  const int nbase = wv * 48;
  for (int k0 = 0; k0 < H_DIM; k0 += 32) {
    v16bf a0 = load_a_frag(&xs[0][0],  H_DIM, k0, lane);
    v16bf a1 = load_a_frag(&xs[16][0], H_DIM, k0, lane);
    if (k0 + 32 < H_DIM)
      __builtin_prefetch(wg_e + (size_t)(k0 + 32) * I_DIM + nbase, 0, 1);
#pragma unroll
    for (int ct = 0; ct < 3; ++ct) {
      const int n0 = nbase + ct * 16;
      stage_b_tile(wg_e + (size_t)k0 * I_DIM + n0, I_DIM, bstage[wv][0], lane);
      stage_b_tile(wu_e + (size_t)k0 * I_DIM + n0, I_DIM, bstage[wv][1], lane);
      v16bf bg = load_b_frag(bstage[wv][0], lane);
      v16bf bu = load_b_frag(bstage[wv][1], lane);
      accg[ct][0] = WMMA_BF16(a0, bg, accg[ct][0]);
      accg[ct][1] = WMMA_BF16(a1, bg, accg[ct][1]);
      accu[ct][0] = WMMA_BF16(a0, bu, accu[ct][0]);
      accu[ct][1] = WMMA_BF16(a1, bu, accu[ct][1]);
    }
  }

  // SiLU(gate) * up -> hs. C layout: VGPR r -> M = r + (lane<16?0:8), N = lane&15.
  {
    const int moff = (lane < 16) ? 0 : 8;
    const int nn   = lane & 15;
#pragma unroll
    for (int ct = 0; ct < 3; ++ct) {
      const int n0 = nbase + ct * 16;
#pragma unroll
      for (int m = 0; m < 2; ++m)
#pragma unroll
        for (int r = 0; r < 8; ++r) {
          float g = accg[ct][m][r], u = accu[ct][m][r];
          float h = (g / (1.f + __expf(-g))) * u;
          hs[m * 16 + r + moff][n0 + nn] = (bf16)h;
        }
    }
  }
  __syncthreads();

  // Stage 2: down projection. Each wave owns 128 of H=2048 columns (8 N-tiles),
  // processed 4 at a time, two M-tiles deep.
  const int moff = (lane < 16) ? 0 : 8;
  const int nn   = lane & 15;
  for (int og = 0; og < 2; ++og) {
    v8f acc[4][2];
#pragma unroll
    for (int tt = 0; tt < 4; ++tt)
#pragma unroll
      for (int m = 0; m < 2; ++m) acc[tt][m] = {};
    for (int k0 = 0; k0 < I_DIM; k0 += 32) {
      v16bf a0 = load_a_frag(&hs[0][0],  I_DIM, k0, lane);
      v16bf a1 = load_a_frag(&hs[16][0], I_DIM, k0, lane);
#pragma unroll
      for (int tt = 0; tt < 4; ++tt) {
        const int n0 = wv * 128 + og * 64 + tt * 16;
        stage_b_tile(wd_e + (size_t)k0 * H_DIM + n0, H_DIM, bstage[wv][tt & 1], lane);
        v16bf b = load_b_frag(bstage[wv][tt & 1], lane);
        acc[tt][0] = WMMA_BF16(a0, b, acc[tt][0]);
        acc[tt][1] = WMMA_BF16(a1, b, acc[tt][1]);
      }
    }
#pragma unroll
    for (int tt = 0; tt < 4; ++tt) {
      const int n0 = wv * 128 + og * 64 + tt * 16;
#pragma unroll
      for (int m = 0; m < 2; ++m)
#pragma unroll
        for (int r = 0; r < 8; ++r) {
          const int M = m * 16 + r + moff;
          atomicAdd(&out[(size_t)sid[M] * H_DIM + n0 + nn], swt[M] * acc[tt][m][r]);
        }
    }
  }
}

extern "C" void kernel_launch(void* const* d_in, const int* in_sizes, int n_in,
                              void* d_out, int out_size, void* d_ws, size_t ws_size,
                              hipStream_t stream) {
  (void)in_sizes; (void)n_in; (void)out_size; (void)ws_size;
  const float* x  = (const float*)d_in[0];
  const float* wr = (const float*)d_in[1];
  const float* wg = (const float*)d_in[2];
  const float* wu = (const float*)d_in[3];
  const float* wd = (const float*)d_in[4];
  float* out = (float*)d_out;

  char*  ws      = (char*)d_ws;
  int*   counts  = (int*)ws;                                       // 64 ints
  int*   tok_ids = (int*)(ws + 256);                               // 64*2048 ints
  float* tok_w   = (float*)(ws + 256 + (size_t)N_EXP * T_TOK * 4); // 64*2048 floats

  (void)hipMemsetAsync(counts, 0, N_EXP * sizeof(int), stream);
  (void)hipMemsetAsync(out, 0, (size_t)T_TOK * H_DIM * sizeof(float), stream);

  router_kernel<<<T_TOK, N_EXP, 0, stream>>>(x, wr, counts, tok_ids, tok_w);
  moe_ffn_kernel<<<dim3(T_TOK / M_TILE, N_EXP), NWAVE * 32, 0, stream>>>(
      x, wg, wu, wd, counts, tok_ids, tok_w, out);
}
